// GAT_classifier_60498909331629
// MI455X (gfx1250) — matrix operand order repaired and checked by reading, not compiled, not run on hardware
//
#include <hip/hip_runtime.h>

// ---------------------------------------------------------------------------
// MI455X (gfx1250) GATv2 classifier.
// bf16 WMMA (v_wmma_f32_16x16x32_bf16) for all GEMMs, fp32 elsewhere.
// Edge softmax/aggregation per-graph in LDS (edges never cross graphs).
// ---------------------------------------------------------------------------

typedef __attribute__((ext_vector_type(16))) __bf16 v16bf;
typedef __attribute__((ext_vector_type(8)))  float  v8f;

#define NNODES   98304   // G*P
#define NGRAPH   4096
#define GP       24      // nodes / graph
#define GE       192     // edges / graph (P*DEG)
#define GTOT     216     // + self loops
#define HDIM     128

// monotonic float<->uint encoding for unsigned-int max atomics
__device__ __forceinline__ unsigned encf(float f) {
  unsigned u = __float_as_uint(f);
  return (u & 0x80000000u) ? ~u : (u | 0x80000000u);
}
__device__ __forceinline__ float decf(unsigned e) {
  unsigned u = (e & 0x80000000u) ? (e & 0x7FFFFFFFu) : ~e;
  return __uint_as_float(u);
}

// ---------------------------------------------------------------------------
// Generic dense layer: Y[nrows,NOUT] = act(X[nrows,K] @ W[K,NOUT] + b)
// REQUIRES nrows % 128 == 0 (true for all call sites).
// Block = 256 threads = 8 waves; each wave owns one 16-row M-tile.
// W staged fp32->bf16 in LDS; A frags from global; D staged through a small
// per-wave LDS tile for coalesced b128 stores (when NOUT % 16 == 0).
// ---------------------------------------------------------------------------
template <int K, int NOUT, bool ACT>
__global__ __launch_bounds__(256) void dense_wmma_kernel(
    const float* __restrict__ X, const float* __restrict__ W,
    const float* __restrict__ bias, float* __restrict__ Y,
    int nrows, float slope)
{
  constexpr int KPAD = (K + 31) & ~31;
  constexpr int NPAD = (NOUT + 15) & ~15;
  constexpr int NKS  = KPAD / 32;
  constexpr int NTIL = NPAD / 16;
  constexpr int WPITCH = NPAD + 8;          // bf16 elems; 272B rows stay 16B-aligned

  __shared__ __align__(16) __bf16 Wlds[KPAD * WPITCH];
  __shared__ __align__(16) float  blds[NPAD];
  __shared__ __align__(16) float  dstage[8][16 * 16];   // per-wave D transpose tile

  const int tid = threadIdx.x;

  // stage weights (bf16) + bias into LDS, zero-padded
  for (int idx = tid; idx < KPAD * NPAD; idx += 256) {
    int k = idx / NPAD, n = idx % NPAD;
    float v = (k < K && n < NOUT) ? W[k * NOUT + n] : 0.f;
    Wlds[k * WPITCH + n] = (__bf16)v;
  }
  for (int n = tid; n < NPAD; n += 256) blds[n] = (n < NOUT) ? bias[n] : 0.f;
  __syncthreads();

  const int wave = tid >> 5, lane = tid & 31;
  const int m = lane & 15, kh = lane >> 4;            // A layout: row, K-octet half
  const int rbase = blockIdx.x * 128 + wave * 16;
  const float* xrow = X + (long)(rbase + m) * K;

  // Build A fragments (ISA 16-bit A 16x32 layout):
  //   element i -> K = 32*ks + 16*(i/8) + 8*kh + (i%8)
  v16bf afrag[NKS];
#pragma unroll
  for (int ks = 0; ks < NKS; ++ks) {
    v16bf a;
#pragma unroll
    for (int i = 0; i < 16; ++i) {
      int k = 32 * ks + 16 * (i >> 3) + 8 * kh + (i & 7);
      float f = (k < K) ? xrow[k] : 0.f;
      a[i] = (__bf16)f;
    }
    afrag[ks] = a;
  }

  float* tp = &dstage[wave][0];
  const int rr = lane >> 1, cc = (lane & 1) * 8;      // readback mapping

#pragma unroll
  for (int nt = 0; nt < NTIL; ++nt) {
    v8f acc = {};
#pragma unroll
    for (int ks = 0; ks < NKS; ++ks) {
      // B layout: column m, element i -> K = 32*ks + 16*(i/8) + 8*kh + (i%8)
      const __bf16* bp = &Wlds[(32 * ks) * WPITCH + nt * 16 + m];
      union { uint4 q[2]; v16bf v; } fb;
#pragma unroll
      for (int i = 0; i < 16; ++i)
        fb.v[i] = bp[(16 * (i >> 3) + 8 * kh + (i & 7)) * WPITCH];
      acc = __builtin_amdgcn_wmma_f32_16x16x32_bf16(
          false, afrag[ks], false, fb.v, (short)0, acc, false, false);
    }

    if constexpr ((NOUT & 15) == 0) {
      // transpose D through per-wave LDS tile, then coalesced b128 stores
      // C/D layout: element i -> M = i + 8*kh, N = m
#pragma unroll
      for (int i = 0; i < 8; ++i) tp[(i + 8 * kh) * 16 + m] = acc[i];
      float ov[8];
      *(float4*)&ov[0] = *(const float4*)&tp[rr * 16 + cc];
      *(float4*)&ov[4] = *(const float4*)&tp[rr * 16 + cc + 4];
#pragma unroll
      for (int q = 0; q < 8; ++q) {
        float v = ov[q] + blds[nt * 16 + cc + q];
        if (ACT) v = v >= 0.f ? v : v * slope;
        ov[q] = v;
      }
      float* yp = Y + (long)(rbase + rr) * NOUT + nt * 16 + cc;
      *(float4*)yp       = *(const float4*)&ov[0];
      *(float4*)(yp + 4) = *(const float4*)&ov[4];
    } else {
      // ragged tail (NOUT=3): guarded scalar stores, col guard only
      const int col = nt * 16 + m;
      const float bv = blds[col];
      if (col < NOUT) {
#pragma unroll
        for (int i = 0; i < 8; ++i) {
          float v = acc[i] + bv;
          if (ACT) v = v >= 0.f ? v : v * slope;
          Y[(long)(rbase + i + 8 * kh) * NOUT + col] = v;
        }
      }
    }
  }
}

// ---------------------------------------------------------------------------
// Per-graph GATv2 attention + aggregation, fully in LDS.
// Block = 256 threads handles one graph's 24 nodes / 216 edges.
// ---------------------------------------------------------------------------
__global__ __launch_bounds__(256) void gat_graph_kernel(
    const float* __restrict__ xl, const float* __restrict__ xr,
    const int* __restrict__ src, const int* __restrict__ dst,
    const float* __restrict__ att, const float* __restrict__ cbias,
    float* __restrict__ hout, int apply_act)
{
  __shared__ float xls[GP][HDIM];
  __shared__ float xrs[GP][HDIM];
  __shared__ float accs[GP][HDIM];
  __shared__ float atts[HDIM], cbs[HDIM];
  __shared__ float logit[GTOT], exs[GTOT];
  __shared__ int   sl[GTOT], dl[GTOT];
  __shared__ unsigned maxe[GP];
  __shared__ float sume[GP];

  const int g = blockIdx.x, tid = threadIdx.x;
  const int nbase = g * GP;

  if (tid < HDIM) { atts[tid] = att[tid]; cbs[tid] = cbias[tid]; }
  for (int idx = tid; idx < GP * HDIM; idx += 256) {
    int n = idx >> 7, f = idx & 127;
    long r = (long)(nbase + n) * HDIM + f;
    xls[n][f] = xl[r];
    xrs[n][f] = xr[r];
    accs[n][f] = 0.f;
  }
  for (int j = tid; j < GTOT; j += 256) {
    if (j < GE) { sl[j] = src[g * GE + j] - nbase; dl[j] = dst[g * GE + j] - nbase; }
    else        { sl[j] = j - GE;                  dl[j] = j - GE; }
  }
  if (tid < GP) { maxe[tid] = 0x007FFFFFu /* encf(-inf) */; sume[tid] = 0.f; }
  __syncthreads();

  const int wave = tid >> 5, lane = tid & 31;
  const int k0 = lane * 4;

  // pass 1: logits = leaky_0.2(xl[s]+xr[d]) . att ; running max per dst
  for (int j = wave; j < GTOT; j += 8) {
    int s = sl[j], d = dl[j];
    float p = 0.f;
#pragma unroll
    for (int q = 0; q < 4; ++q) {
      float v = xls[s][k0 + q] + xrs[d][k0 + q];
      v = v >= 0.f ? v : v * 0.2f;
      p += v * atts[k0 + q];
    }
#pragma unroll
    for (int off = 16; off; off >>= 1) p += __shfl_xor(p, off, 32);
    if (lane == 0) { logit[j] = p; atomicMax(&maxe[d], encf(p)); }
  }
  __syncthreads();

  // pass 2: stable exp + per-dst sum (ds_add_f32)
  for (int j = tid; j < GTOT; j += 256) {
    int d = dl[j];
    float e = __expf(logit[j] - decf(maxe[d]));
    exs[j] = e;
    atomicAdd(&sume[d], e);
  }
  __syncthreads();

  // pass 3: out[d] += alpha * xl[s]
  for (int j = wave; j < GTOT; j += 8) {
    int s = sl[j], d = dl[j];
    float coef = exs[j] / sume[d];
#pragma unroll
    for (int q = 0; q < 4; ++q)
      atomicAdd(&accs[d][k0 + q], xls[s][k0 + q] * coef);
  }
  __syncthreads();

  // bias + optional inter-layer leaky, write new h
  for (int idx = tid; idx < GP * HDIM; idx += 256) {
    int n = idx >> 7, f = idx & 127;
    float v = accs[n][f] + cbs[f];
    if (apply_act) v = v >= 0.f ? v : v * 0.01f;
    hout[(long)(nbase + n) * HDIM + f] = v;
  }
}

// ---------------------------------------------------------------------------
// global_add_pool + concat(extra) -> z[G, 136]
// ---------------------------------------------------------------------------
__global__ __launch_bounds__(160) void pool_kernel(
    const float* __restrict__ h, const float* __restrict__ extra,
    float* __restrict__ z)
{
  const int g = blockIdx.x, f = threadIdx.x;
  if (f < HDIM) {
    float s = 0.f;
#pragma unroll 4
    for (int p = 0; p < GP; ++p) s += h[(long)(g * GP + p) * HDIM + f];
    z[g * 136 + f] = s;
  } else if (f < 136) {
    z[g * 136 + f] = extra[g * 8 + (f - HDIM)];
  }
}

// ---------------------------------------------------------------------------
extern "C" void kernel_launch(void* const* d_in, const int* in_sizes, int n_in,
                              void* d_out, int out_size, void* d_ws, size_t ws_size,
                              hipStream_t stream)
{
  const float* x     = (const float*)d_in[0];
  const int*   src   = (const int*)  d_in[1];
  const int*   dst   = (const int*)  d_in[2];
  // d_in[3] = batch (graphs are contiguous blocks of 24 nodes; not needed)
  const float* extra = (const float*)d_in[4];
  const float* Wi0   = (const float*)d_in[5];
  const float* bi0   = (const float*)d_in[6];
  const float* Wi1   = (const float*)d_in[7];
  const float* bi1   = (const float*)d_in[8];
  const float* Wl    = (const float*)d_in[9];
  const float* bl    = (const float*)d_in[10];
  const float* Wr    = (const float*)d_in[11];
  const float* br    = (const float*)d_in[12];
  const float* att   = (const float*)d_in[13];
  const float* cb    = (const float*)d_in[14];
  const float* Wf0   = (const float*)d_in[15];
  const float* bf0   = (const float*)d_in[16];
  const float* Wf1   = (const float*)d_in[17];
  const float* bf1   = (const float*)d_in[18];
  const float* Wf2   = (const float*)d_in[19];
  const float* bf2   = (const float*)d_in[20];

  // scratch layout (floats): two N x 128 feature buffers + small tails
  float* buf0 = (float*)d_ws;                       // h / xr / new h
  float* buf1 = buf0 + (size_t)NNODES * HDIM;       // xl / init tmp
  float* z    = buf1 + (size_t)NNODES * HDIM;       // [G,136]
  float* z1   = z    + (size_t)NGRAPH * 136;        // [G,128]
  float* z2   = z1   + (size_t)NGRAPH * HDIM;       // [G,128]

  const dim3 blk(256);
  const dim3 gN(NNODES / 128);    // 768
  const dim3 gG(NGRAPH / 128);    // 32

  // init MLP: x[.,32] -> buf1 -> buf0
  dense_wmma_kernel<32, 128, true><<<gN, blk, 0, stream>>>(x, Wi0, bi0, buf1, NNODES, 0.01f);
  dense_wmma_kernel<128, 128, true><<<gN, blk, 0, stream>>>(buf1, Wi1, bi1, buf0, NNODES, 0.01f);

  // 6 GATv2 layers (2 stacks x 3 message passes)
  for (int c = 0; c < 6; ++c) {
    const float* wl = Wl + (size_t)c * HDIM * HDIM;
    const float* wr = Wr + (size_t)c * HDIM * HDIM;
    // xl -> buf1 ; xr -> buf0 (safe: each block stages its rows before rewriting)
    dense_wmma_kernel<128, 128, false><<<gN, blk, 0, stream>>>(buf0, wl, bl + c * HDIM, buf1, NNODES, 0.f);
    dense_wmma_kernel<128, 128, false><<<gN, blk, 0, stream>>>(buf0, wr, br + c * HDIM, buf0, NNODES, 0.f);
    int apply_act = ((c % 3) < 2) ? 1 : 0;   // act between convs, not after last of each stack
    gat_graph_kernel<<<NGRAPH, blk, 0, stream>>>(buf1, buf0, src, dst,
                                                 att + c * HDIM, cb + c * HDIM,
                                                 buf0, apply_act);
  }

  // pool + final MLP
  pool_kernel<<<NGRAPH, dim3(160), 0, stream>>>(buf0, extra, z);
  dense_wmma_kernel<136, 128, true><<<gG, blk, 0, stream>>>(z,  Wf0, bf0, z1, NGRAPH, 0.01f);
  dense_wmma_kernel<128, 128, true><<<gG, blk, 0, stream>>>(z1, Wf1, bf1, z2, NGRAPH, 0.01f);
  dense_wmma_kernel<128, 3, false><<<gG, blk, 0, stream>>>(z2, Wf2, bf2, (float*)d_out, NGRAPH, 0.f);
}
